// UaNet_21792664060233
// MI455X (gfx1250) — compile-verified, matrix operand order checked by reading, not run on hardware
//
#include <hip/hip_runtime.h>
#include <math.h>

// ---------------------------------------------------------------------------
// UaNet-style 3D detector forward pass for gfx1250 (MI455X), wave32 + WMMA.
// Conv3D / FC as implicit GEMM on v_wmma_f32_16x16x32_f16:
//   - MT (2|4) 16-row M tiles per block, 64 N-voxels (4 waves x 16)
//   - LDS double-buffered, 1 barrier per 32-deep K step
//   - staging is branch-free and PAIRED (one ds_store_b32 per K-pair), and is
//     split into a load phase (all gathers in flight) + convert/store phase so
//     a single loadcnt drain covers the whole slab instead of per-load waits
// BN(eval)+bias+residual+ReLU fused into the WMMA epilogue.
// ---------------------------------------------------------------------------

typedef __attribute__((ext_vector_type(16))) _Float16 v16h;
typedef __attribute__((ext_vector_type(2)))  _Float16 v2h;
typedef __attribute__((ext_vector_type(8)))  float    v8f;

#define BNS 0.9999950000374997f   /* 1/sqrt(1+1e-5), eval-mode BN */

// K-index of LDS slot (lane ls, element e) per CDNA5 ISA 7.12.2.
// For e = 2u, 2u+1 the K indices are consecutive (pair-store friendly).
__device__ __forceinline__ int wmma_k_of(int ls, int e) {
    return (e & 7) + ((e >> 3) << 4) + ((ls >> 4) << 3);
}

// ---------------------------------------------------------------------------
// Implicit-GEMM 3D conv. KS = kernel size (3|1), MT = # of 16-row M tiles.
// Block: 128 threads = 4 waves. Tile: (16*MT) cout x 64 voxels.
// ---------------------------------------------------------------------------
template <int KS, int MT>
__global__ __launch_bounds__(128)
void conv3d_wmma_k(const float* __restrict__ in,
                   const float* __restrict__ wgt,   // [cout, cin*KS^3]
                   const float* __restrict__ bias,  // [cout]
                   const float* __restrict__ gam,   // [cout]
                   const float* __restrict__ bet,   // [cout]
                   const float* __restrict__ res,   // [cout, nvox] or null
                   float* __restrict__ out,         // [cout, nvox]
                   int cin, int cout,
                   int inD, int inH, int inW,
                   int oD, int oH, int oW,
                   int stride, int pad, int relu)
{
    __shared__ alignas(32) _Float16 lA[2][MT][32][16];  // [buf][Mtile][lane][e]
    __shared__ alignas(32) _Float16 lB[2][4][32][16];   // [buf][wave][lane][e]

    const int tid  = threadIdx.x;
    const int lane = tid & 31;
    const int wv   = tid >> 5;

    const int nvox    = oD * oH * oW;
    const int voxBase = blockIdx.x * 64;
    const int coBase  = blockIdx.y * (16 * MT);

    constexpr int kvol = KS * KS * KS;
    const int Ktot   = cin * kvol;
    const int inHW   = inH * inW;
    const int nsteps = (Ktot + 31) >> 5;

    // ---- hoisted invariants: A pairs (2*MT per thread) ----
    int awo[2 * MT], aok[2 * MT], ako[2 * MT];
#pragma unroll
    for (int j = 0; j < 2 * MT; ++j) {
        int pid = tid + (j << 7);          // 0 .. MT*256-1
        int mt  = pid >> 8;
        int r   = pid & 255;
        int ls  = r >> 3, u = r & 7;
        int co  = coBase + (mt << 4) + (ls & 15);
        aok[j] = (co < cout);
        awo[j] = (co < cout ? co : 0) * Ktot;
        ako[j] = wmma_k_of(ls, 2 * u);
    }

    // ---- hoisted invariants: B pairs (8 per thread); coords bit-packed ----
    int bcp[8], bko[8];
#pragma unroll
    for (int j = 0; j < 8; ++j) {
        int pid = tid + (j << 7);          // 0..1023
        int w8  = pid >> 8;
        int r   = pid & 255;
        int ls  = r >> 3, u = r & 7;
        int vox = voxBase + (w8 << 4) + (ls & 15);
        int vc  = vox < nvox ? vox : 0;
        int ox  = vc % oW;
        int t2  = vc / oW;
        int oy  = t2 % oH;
        int oz  = t2 / oH;
        int vz = oz * stride - pad, vy = oy * stride - pad, vx = ox * stride - pad;
        bcp[j] = ((vz + 2) << 20) | ((vy + 2) << 10) | (vx + 2) | ((vox < nvox) << 30);
        bko[j] = wmma_k_of(ls, 2 * u);
    }

    // im2col gather index + validity for one K element.
    auto gidx = [&](int k, int vz, int vy, int vx, int vok, int& ok) -> int {
        int kc = k < Ktot ? k : 0;
        int ci = kc / kvol;                  // constant divisor -> mul/shift
        int tap = kc - ci * kvol;
        int dz = tap / (KS * KS);
        int t9 = tap - dz * (KS * KS);
        int dy = t9 / KS;
        int dx = t9 - dy * KS;
        int iz = vz + dz, iy = vy + dy, ix = vx + dx;
        ok = vok & (k < Ktot) &
             ((unsigned)iz < (unsigned)inD) &
             ((unsigned)iy < (unsigned)inH) &
             ((unsigned)ix < (unsigned)inW);
        int idx = (ci * inD + iz) * inHW + iy * inW + ix;
        return ok ? idx : 0;
    };

    auto stage = [&](int kk, int nb) {
        __builtin_prefetch(&wgt[awo[0] + (kk + 32 < Ktot ? kk + 32 : 0)], 0, 1);

        // ---------- load phase: issue ALL gathers, no consumption ----------
        float av0[2 * MT], av1[2 * MT];
#pragma unroll
        for (int j = 0; j < 2 * MT; ++j) {
            int k0  = kk + ako[j];
            int ok0 = aok[j] & (k0 < Ktot);
            int ok1 = aok[j] & (k0 + 1 < Ktot);
            av0[j] = wgt[awo[j] + (ok0 ? k0 : 0)];
            av1[j] = wgt[awo[j] + (ok1 ? k0 + 1 : 0)];
        }
        float bv0[8], bv1[8];
        int   bm[8];
#pragma unroll
        for (int j = 0; j < 8; ++j) {
            int vz = ((bcp[j] >> 20) & 1023) - 2;
            int vy = ((bcp[j] >> 10) & 1023) - 2;
            int vx = (bcp[j] & 1023) - 2;
            int vok = (bcp[j] >> 30) & 1;
            int k0 = kk + bko[j];
            int ok0, ok1;
            int i0 = gidx(k0,     vz, vy, vx, vok, ok0);
            int i1 = gidx(k0 + 1, vz, vy, vx, vok, ok1);
            bv0[j] = in[i0];
            bv1[j] = in[i1];
            bm[j]  = ok0 | (ok1 << 1);
        }

        // ---------- convert/store phase ----------
#pragma unroll
        for (int j = 0; j < 2 * MT; ++j) {
            int pid = tid + (j << 7);
            int mt  = pid >> 8;
            int r   = pid & 255;
            int ls  = r >> 3, u = r & 7;
            int k0  = kk + ako[j];
            int ok0 = aok[j] & (k0 < Ktot);
            int ok1 = aok[j] & (k0 + 1 < Ktot);
            v2h p;
            p[0] = ok0 ? (_Float16)av0[j] : (_Float16)0.0f;
            p[1] = ok1 ? (_Float16)av1[j] : (_Float16)0.0f;
            *(v2h*)(&lA[nb][mt][ls][2 * u]) = p;
        }
#pragma unroll
        for (int j = 0; j < 8; ++j) {
            int pid = tid + (j << 7);
            int w8  = pid >> 8;
            int r   = pid & 255;
            int ls  = r >> 3, u = r & 7;
            v2h p;
            p[0] = (bm[j] & 1) ? (_Float16)bv0[j] : (_Float16)0.0f;
            p[1] = (bm[j] & 2) ? (_Float16)bv1[j] : (_Float16)0.0f;
            *(v2h*)(&lB[nb][w8][ls][2 * u]) = p;
        }
    };

    v8f acc[MT];
    v8f z = {};
#pragma unroll
    for (int mt = 0; mt < MT; ++mt) acc[mt] = z;

    stage(0, 0);
    for (int it = 0; it < nsteps; ++it) {
        __syncthreads();
        int cb = it & 1;
        if (it + 1 < nsteps) stage((it + 1) << 5, cb ^ 1);
        v16h b = *(const v16h*)(&lB[cb][wv][lane][0]);
#pragma unroll
        for (int mt = 0; mt < MT; ++mt) {
            v16h a = *(const v16h*)(&lA[cb][mt][lane][0]);
            acc[mt] = __builtin_amdgcn_wmma_f32_16x16x32_f16(
                          false, a, false, b, (short)0, acc[mt], false, false);
        }
    }

    // Epilogue: C/D layout — VGPR r, lane l -> M = r + 8*(l>>4), N = l&15.
    const int n   = lane & 15;
    const int mh  = lane >> 4;
    const int vox = voxBase + (wv << 4) + n;
    if (vox < nvox) {
#pragma unroll
        for (int mt = 0; mt < MT; ++mt) {
#pragma unroll
            for (int r = 0; r < 8; ++r) {
                int co = coBase + (mt << 4) + r + (mh << 3);
                if (co < cout) {
                    float v = acc[mt][r] + bias[co];
                    v = v * (gam[co] * BNS) + bet[co];
                    if (res) v += res[co * nvox + vox];
                    if (relu) v = fmaxf(v, 0.0f);
                    out[co * nvox + vox] = v;
                }
            }
        }
    }
}

// ---------------------------------------------------------------------------
// Dense GEMM (FC head): out[m, colOff+n] = act( A[m,:] @ W[:,n] + bias[n] )
// Tile: 64 rows x 64 cols per block. K multiple of 32 assumed (8192/512/256),
// so A pairs are one aligned float2 load. Load / store phases split.
// ---------------------------------------------------------------------------
__global__ __launch_bounds__(128)
void gemm_wmma_k(const float* __restrict__ A,
                 const float* __restrict__ W,
                 const float* __restrict__ bias,
                 float* __restrict__ out,
                 int M, int N, int K, int ldout, int colOff, int relu)
{
    constexpr int MT = 4;
    __shared__ alignas(32) _Float16 lA[2][MT][32][16];
    __shared__ alignas(32) _Float16 lB[2][4][32][16];

    const int tid  = threadIdx.x;
    const int lane = tid & 31;
    const int wv   = tid >> 5;

    const int rowBase = blockIdx.y * (16 * MT);
    const int colBase = blockIdx.x * 64;
    const int nsteps  = (K + 31) >> 5;

    int aro[2 * MT], aok[2 * MT], ako[2 * MT];
#pragma unroll
    for (int j = 0; j < 2 * MT; ++j) {
        int pid = tid + (j << 7);
        int mt  = pid >> 8;
        int r   = pid & 255;
        int ls  = r >> 3, u = r & 7;
        int m   = rowBase + (mt << 4) + (ls & 15);
        aok[j] = (m < M);
        aro[j] = (m < M ? m : 0) * K;
        ako[j] = wmma_k_of(ls, 2 * u);
    }
    int bco[8], bok[8], bko[8];
#pragma unroll
    for (int j = 0; j < 8; ++j) {
        int pid = tid + (j << 7);
        int w8  = pid >> 8;
        int r   = pid & 255;
        int ls  = r >> 3, u = r & 7;
        int c   = colBase + (w8 << 4) + (ls & 15);
        bok[j] = (c < N);
        bco[j] = (c < N ? c : 0);
        bko[j] = wmma_k_of(ls, 2 * u);
    }

    auto stage = [&](int kk, int nb) {
        // ---------- load phase ----------
        float2 aw[2 * MT];
#pragma unroll
        for (int j = 0; j < 2 * MT; ++j) {
            int k0 = kk + ako[j];
            int ok = aok[j] & (k0 < K);    // K even, k0 even -> k0+1 < K too
            aw[j] = *(const float2*)(A + aro[j] + (ok ? k0 : 0));
        }
        float bv0[8], bv1[8];
#pragma unroll
        for (int j = 0; j < 8; ++j) {
            int k0 = kk + bko[j];
            int ok = bok[j] & (k0 < K);
            bv0[j] = W[(ok ? k0 : 0) * N + bco[j]];
            bv1[j] = W[(ok ? k0 + 1 : 0) * N + bco[j]];
        }
        // ---------- convert/store phase ----------
#pragma unroll
        for (int j = 0; j < 2 * MT; ++j) {
            int pid = tid + (j << 7);
            int mt  = pid >> 8;
            int r   = pid & 255;
            int ls  = r >> 3, u = r & 7;
            int k0  = kk + ako[j];
            int ok  = aok[j] & (k0 < K);
            v2h p;
            p[0] = ok ? (_Float16)aw[j].x : (_Float16)0.0f;
            p[1] = ok ? (_Float16)aw[j].y : (_Float16)0.0f;
            *(v2h*)(&lA[nb][mt][ls][2 * u]) = p;
        }
#pragma unroll
        for (int j = 0; j < 8; ++j) {
            int pid = tid + (j << 7);
            int w8  = pid >> 8;
            int r   = pid & 255;
            int ls  = r >> 3, u = r & 7;
            int k0  = kk + bko[j];
            int ok  = bok[j] & (k0 < K);
            v2h p;
            p[0] = ok ? (_Float16)bv0[j] : (_Float16)0.0f;
            p[1] = ok ? (_Float16)bv1[j] : (_Float16)0.0f;
            *(v2h*)(&lB[nb][w8][ls][2 * u]) = p;
        }
    };

    v8f acc[MT];
    v8f z = {};
#pragma unroll
    for (int mt = 0; mt < MT; ++mt) acc[mt] = z;

    stage(0, 0);
    for (int it = 0; it < nsteps; ++it) {
        __syncthreads();
        int cb = it & 1;
        if (it + 1 < nsteps) stage((it + 1) << 5, cb ^ 1);
        v16h b = *(const v16h*)(&lB[cb][wv][lane][0]);
#pragma unroll
        for (int mt = 0; mt < MT; ++mt) {
            v16h a = *(const v16h*)(&lA[cb][mt][lane][0]);
            acc[mt] = __builtin_amdgcn_wmma_f32_16x16x32_f16(
                          false, a, false, b, (short)0, acc[mt], false, false);
        }
    }

    const int n  = colBase + (wv << 4) + (lane & 15);
    const int mh = lane >> 4;
    if (n < N) {
        float bv = bias[n];
#pragma unroll
        for (int mt = 0; mt < MT; ++mt) {
#pragma unroll
            for (int r = 0; r < 8; ++r) {
                int m = rowBase + (mt << 4) + r + (mh << 3);
                if (m < M) {
                    float v = acc[mt][r] + bv;
                    if (relu) v = fmaxf(v, 0.0f);
                    out[m * ldout + colOff + n] = v;
                }
            }
        }
    }
}

// ---------------------------------------------------------------------------
// 2x2x2 max pool
// ---------------------------------------------------------------------------
__global__ void maxpool2_k(const float* __restrict__ in, float* __restrict__ out,
                           int C, int oD, int oH, int oW)
{
    const int iH = oH * 2, iW = oW * 2;
    const long n = (long)C * oD * oH * oW;
    for (long i = (long)blockIdx.x * blockDim.x + threadIdx.x; i < n;
         i += (long)gridDim.x * blockDim.x) {
        int x = (int)(i % oW);
        long t = i / oW;
        int y = (int)(t % oH); t /= oH;
        int z = (int)(t % oD);
        int c = (int)(t / oD);
        const float* base = in + (((long)c * (oD * 2) + 2 * z) * iH + 2 * y) * iW + 2 * x;
        float m = fmaxf(fmaxf(base[0], base[1]), fmaxf(base[iW], base[iW + 1]));
        const float* b2 = base + (long)iH * iW;
        m = fmaxf(m, fmaxf(fmaxf(b2[0], b2[1]), fmaxf(b2[iW], b2[iW + 1])));
        out[i] = m;
    }
}

// ---------------------------------------------------------------------------
// ConvTranspose3d(kernel=2, stride=2) + BN + ReLU.  w: (cin, cout, 2,2,2)
// ---------------------------------------------------------------------------
__global__ void deconv2_k(const float* __restrict__ in, const float* __restrict__ w,
                          const float* __restrict__ bias,
                          const float* __restrict__ gam, const float* __restrict__ bet,
                          float* __restrict__ out,
                          int cin, int cout, int iD, int iH, int iW)
{
    const int oD = 2 * iD, oH = 2 * iH, oW = 2 * iW;
    const long n = (long)cout * oD * oH * oW;
    for (long i = (long)blockIdx.x * blockDim.x + threadIdx.x; i < n;
         i += (long)gridDim.x * blockDim.x) {
        int ox = (int)(i % oW);
        long t = i / oW;
        int oy = (int)(t % oH); t /= oH;
        int oz = (int)(t % oD);
        int co = (int)(t / oD);
        int d = oz >> 1, p = oz & 1;
        int h = oy >> 1, q = oy & 1;
        int wd = ox >> 1, r = ox & 1;
        float acc = bias[co];
        for (int ci = 0; ci < cin; ++ci) {
            float xv = in[((long)ci * iD + d) * iH * iW + (long)h * iW + wd];
            float wv = w[(((long)ci * cout + co) * 2 + p) * 4 + q * 2 + r];
            acc = fmaf(xv, wv, acc);
        }
        float v = acc * (gam[co] * BNS) + bet[co];
        out[i] = fmaxf(v, 0.0f);
    }
}

// ---------------------------------------------------------------------------
// CropRoi (fixed 8^3 crop at stride 8) + adaptive max pool 8->4 (2x2x2 pool).
// ---------------------------------------------------------------------------
__global__ void crop_pool_k(const float* __restrict__ feat,
                            const float* __restrict__ props,
                            float* __restrict__ crops,
                            int C, int D, int H, int W)
{
    const int p = blockIdx.x;
    const float* pr = props + (long)p * 8;
    int c0z = (int)floorf((pr[2] - pr[5] * 0.5f) * 0.125f);
    int c0y = (int)floorf((pr[3] - pr[6] * 0.5f) * 0.125f);
    int c0x = (int)floorf((pr[4] - pr[7] * 0.5f) * 0.125f);
    c0z = min(max(c0z, 0), D - 8);
    c0y = min(max(c0y, 0), H - 8);
    c0x = min(max(c0x, 0), W - 8);

    for (int idx = threadIdx.x; idx < C * 64; idx += blockDim.x) {
        int c = idx >> 6;
        int v = idx & 63;
        int pz = v >> 4, py = (v >> 2) & 3, px = v & 3;
        float m = -3.4e38f;
#pragma unroll
        for (int a = 0; a < 2; ++a)
#pragma unroll
            for (int b = 0; b < 2; ++b)
#pragma unroll
                for (int e = 0; e < 2; ++e) {
                    int z = c0z + 2 * pz + a;
                    int y = c0y + 2 * py + b;
                    int x = c0x + 2 * px + e;
                    m = fmaxf(m, feat[((long)c * D + z) * H * W + (long)y * W + x]);
                }
        crops[(long)p * C * 64 + idx] = m;
    }
}

// ---------------------------------------------------------------------------
// Host orchestration
// ---------------------------------------------------------------------------
static inline int cdiv(int a, int b) { return (a + b - 1) / b; }

extern "C" void kernel_launch(void* const* d_in, const int* in_sizes, int n_in,
                              void* d_out, int out_size, void* d_ws, size_t ws_size,
                              hipStream_t stream)
{
    (void)in_sizes; (void)n_in; (void)out_size; (void)ws_size;
    const float* X     = (const float*)d_in[0];   // (1,1,96,128,128)
    const float* PROPS = (const float*)d_in[1];   // (128, 8)

    // params flattened jax-tree style (nested dict keys sorted), leaves from d_in[2]:
    //  back3[0]@2 back3[1]@10 back3[2]@18 (b1,b2,be1,be2,g1,g2,w1,w2)
    //  delta_b=26 delta_w=27 fc1_b=28 fc1_w=29 fc2_b=30 fc2_w=31
    //  forw1[0]@32 (b1,b2,be1,be2,bes,bs,g1,g2,gs,w1,w2,ws) forw1[1]@44
    //  forw2[0]@52 forw2[1]@64  forw3@72,80,88  forw4@96,104,112
    //  logit_b=120 logit_w=121  path1: b=122 be=123 g=124 w=125
    //  pre: b1=126 b2=127 be1=128 be2=129 g1=130 g2=131 w1=132 w2=133
    auto P = [&](int i) -> const float* { return (const float*)d_in[i]; };

    auto WS = [&](size_t off) -> float* { return (float*)((char*)d_ws + off); };
    float* bufA = WS(0);               // 25.2 MB each
    float* bufB = WS(25165824);
    float* bufC = WS(50331648);
    float* T0   = WS(75497472);        // 2 MB each (/8 scale & below)
    float* T1   = WS(77594624);
    float* T2   = WS(79691776);
    float* CR   = WS(81788928);        // crops 128*8192 f32
    float* H1   = WS(85983232);        // 128*512
    float* H2   = WS(86245376);        // 128*256
    float* OUT  = (float*)d_out;       // 128*196

    dim3 blk(128);
    auto conv = [&](const float* in, int wi, int bi, int gi, int bei,
                    const float* res, float* out,
                    int cin, int cout, int iD, int iH, int iW,
                    int oD, int oH, int oW, int ks, int st, int pad, int relu) {
        bool big = (cout >= 64);
        int mt = big ? 4 : 2;
        dim3 grid(cdiv(oD * oH * oW, 64), cdiv(cout, 16 * mt));
        if (ks == 3) {
            if (big)
                conv3d_wmma_k<3, 4><<<grid, blk, 0, stream>>>(
                    in, P(wi), P(bi), P(gi), P(bei), res, out,
                    cin, cout, iD, iH, iW, oD, oH, oW, st, pad, relu);
            else
                conv3d_wmma_k<3, 2><<<grid, blk, 0, stream>>>(
                    in, P(wi), P(bi), P(gi), P(bei), res, out,
                    cin, cout, iD, iH, iW, oD, oH, oW, st, pad, relu);
        } else {
            if (big)
                conv3d_wmma_k<1, 4><<<grid, blk, 0, stream>>>(
                    in, P(wi), P(bi), P(gi), P(bei), res, out,
                    cin, cout, iD, iH, iW, oD, oH, oW, st, pad, relu);
            else
                conv3d_wmma_k<1, 2><<<grid, blk, 0, stream>>>(
                    in, P(wi), P(bi), P(gi), P(bei), res, out,
                    cin, cout, iD, iH, iW, oD, oH, oW, st, pad, relu);
        }
    };
    auto pool = [&](const float* in, float* out, int C, int oD, int oH, int oW) {
        long n = (long)C * oD * oH * oW;
        maxpool2_k<<<dim3((unsigned)((n + 255) / 256)), dim3(256), 0, stream>>>(
            in, out, C, oD, oH, oW);
    };
    auto gemm = [&](const float* A, int wi, int bi, float* out,
                    int M, int N, int K, int ld, int co, int relu) {
        dim3 grid(cdiv(N, 64), cdiv(M, 64));
        gemm_wmma_k<<<grid, blk, 0, stream>>>(A, P(wi), P(bi), out, M, N, K, ld, co, relu);
    };
    auto res_ns = [&](int b, const float* x, float* h, float* out,
                      int C, int D, int H, int W) {
        conv(x, b + 6, b + 0, b + 4, b + 2, nullptr, h, C, C, D, H, W, D, H, W, 3, 1, 1, 1);
        conv(h, b + 7, b + 1, b + 5, b + 3, x, out, C, C, D, H, W, D, H, W, 3, 1, 1, 1);
    };
    auto res_sc = [&](int b, const float* x, float* sc, float* h, float* out,
                      int cin, int cout, int D, int H, int W) {
        conv(x, b + 11, b + 5, b + 8, b + 4, nullptr, sc, cin, cout, D, H, W, D, H, W, 1, 1, 0, 0);
        conv(x, b + 9,  b + 0, b + 6, b + 2, nullptr, h,  cin, cout, D, H, W, D, H, W, 3, 1, 1, 1);
        conv(h, b + 10, b + 1, b + 7, b + 3, sc, out, cout, cout, D, H, W, D, H, W, 3, 1, 1, 1);
    };

    // ---- preBlock: 1->24 stride-2 conv, then 24->24 ----
    conv(X,    132, 126, 130, 128, nullptr, bufA, 1, 24, 96, 128, 128, 48, 64, 64, 3, 2, 1, 1);
    conv(bufA, 133, 127, 131, 129, nullptr, bufB, 24, 24, 48, 64, 64, 48, 64, 64, 3, 1, 1, 1);

    // ---- forw1 (24->32, 32->32) at /2 ----
    res_sc(32, bufB, bufC, bufA, bufB, 24, 32, 48, 64, 64);  // -> bufB
    res_ns(44, bufB, bufA, bufC, 32, 48, 64, 64);            // -> bufC

    // ---- pool + forw2 (32->64, 64->64) at /4 ----
    pool(bufC, bufA, 32, 24, 32, 32);
    res_sc(52, bufA, bufB, bufC, bufA, 32, 64, 24, 32, 32);  // -> bufA
    res_ns(64, bufA, bufB, bufC, 64, 24, 32, 32);            // -> bufC

    // ---- pool + forw3 (64->64 x3) at /8 ----
    pool(bufC, T0, 64, 12, 16, 16);
    res_ns(72, T0, T1, T2, 64, 12, 16, 16);
    res_ns(80, T2, T1, T0, 64, 12, 16, 16);
    res_ns(88, T0, T1, T2, 64, 12, 16, 16);                  // out3 -> T2

    // ---- pool + forw4 (64->64 x3) at /16 ----
    pool(T2, T0, 64, 6, 8, 8);
    res_ns(96,  T0, T1, bufB, 64, 6, 8, 8);
    res_ns(104, bufB, T1, T0, 64, 6, 8, 8);
    res_ns(112, T0, T1, bufB, 64, 6, 8, 8);                  // out4 -> bufB

    // ---- path1 deconv + BN + ReLU -> comb3 channels [0,64) in bufA ----
    {
        long n = (long)64 * 12 * 16 * 16;
        deconv2_k<<<dim3((unsigned)((n + 255) / 256)), dim3(256), 0, stream>>>(
            bufB, P(125), P(122), P(124), P(123), bufA, 64, 64, 6, 8, 8);
    }
    // concat: out3 -> comb3 channels [64,128)
    hipMemcpyAsync(bufA + (long)64 * 12 * 16 * 16, T2,
                   sizeof(float) * 64 * 12 * 16 * 16, hipMemcpyDeviceToDevice, stream);

    // ---- back3 (128->128 x3) at /8 ----
    res_ns(2,  bufA, bufC, bufB, 128, 12, 16, 16);
    res_ns(10, bufB, bufC, bufA, 128, 12, 16, 16);
    res_ns(18, bufA, bufC, bufB, 128, 12, 16, 16);           // feat -> bufB

    // ---- CropRoi + adaptive max pool (8 -> 4) ----
    crop_pool_k<<<dim3(128), dim3(128), 0, stream>>>(bufB, PROPS, CR, 128, 12, 16, 16);

    // ---- FC head (WMMA GEMMs, bias+ReLU fused) ----
    gemm(CR, 29, 28, H1, 128, 512, 8192, 512, 0, 1);    // fc1 + relu
    gemm(H1, 31, 30, H2, 128, 256,  512, 256, 0, 1);    // fc2 + relu
    gemm(H2, 121, 120, OUT, 128,  28, 256, 196, 0,  0); // logits -> cols [0,28)
    gemm(H2,  27,  26, OUT, 128, 168, 256, 196, 28, 0); // deltas -> cols [28,196)
}
